// rnn_single_50070728737109
// MI455X (gfx1250) — matrix operand order repaired and checked by reading, not compiled, not run on hardware
//
#include <hip/hip_runtime.h>
#include <hip/hip_bf16.h>

// ---------------------------------------------------------------------------
// Types for CDNA5 WMMA (wave32): v_wmma_f32_16x16x32_bf16
// ---------------------------------------------------------------------------
typedef __bf16 bf16_t;
typedef __attribute__((ext_vector_type(16))) __bf16 v16bf;
typedef __attribute__((ext_vector_type(8)))  float  v8f;

union FragBF { v16bf v; uint4 q[2]; };

#define BAG   512
#define ENC   512
#define ATT   512
#define NDIMS 1024
#define KCAT  1536   // ENC + NDIMS for the fused fc1/fc2 GEMM

// ---------------------------------------------------------------------------
// Generic bf16 WMMA GEMM:  C(M,N) = A(M,K) @ B(K,N) [+bias0 +bias1] [relu]
//   A  : bf16 row-major, leading dim lda
//   BT : bf16, B transposed (N x K) row-major, leading dim ldb   (K-major!)
//   C  : f32 row-major, leading dim ldc
// Block: 256 thr = 8 waves as 4(M) x 2(N) grid of 16x16 WMMA tiles -> 64x32.
//
// Fragment layouts per CDNA5 ISA 7.12.2 (wave32):
//   A 16x32 bf16 : lane L holds row m=L&15; half=L>>4 selects K base;
//                  elems 0..7 = K[kb..kb+7], elems 8..15 = K[kb+16..kb+23]
//   B 32x16 bf16 : lane L holds col n=L&15; half selects K base 0/8;
//                  elems 0..7 = K[kb..kb+7], elems 8..15 = K[kb+16..kb+23] (BT row)
//   C/D 16x16 f32: lane L col n=L&15; VGPR r -> row (half*8 + r)
// ---------------------------------------------------------------------------
__global__ __launch_bounds__(256)
void gemm_bf16_wmma(const bf16_t* __restrict__ A, int lda,
                    const bf16_t* __restrict__ BT, int ldb,
                    float* __restrict__ C, int ldc,
                    int K,
                    const float* __restrict__ bias0,
                    const float* __restrict__ bias1,
                    int do_relu)
{
    const int tid   = threadIdx.x;
    const int lane  = tid & 31;
    const int wave  = tid >> 5;
    const int waveM = wave & 3;                // 0..3
    const int waveN = wave >> 2;               // 0..1
    const int tileM = blockIdx.y * 64 + waveM * 16;
    const int tileN = blockIdx.x * 32 + waveN * 16;

    const int half   = lane >> 4;              // 0 or 1
    const int nlocal = lane & 15;
    const int arow   = tileM + nlocal;         // A row held by this lane
    const int bcol   = tileN + nlocal;         // B column (BT row)
    const int kbA    = half * 8;
    const int kbB    = half * 8;

    const bf16_t* aptr = A  + (size_t)arow * lda;
    const bf16_t* bptr = BT + (size_t)bcol * ldb;

    v8f acc = {};
    for (int k0 = 0; k0 < K; k0 += 32) {
        FragBF fa, fb;
        fa.q[0] = *(const uint4*)(aptr + k0 + kbA);        // K = kb..kb+7
        fa.q[1] = *(const uint4*)(aptr + k0 + kbA + 16);   // K = kb+16..kb+23
        fb.q[0] = *(const uint4*)(bptr + k0 + kbB);        // K = kb..kb+7
        fb.q[1] = *(const uint4*)(bptr + k0 + kbB + 16);   // K = kb+16..kb+23
        acc = __builtin_amdgcn_wmma_f32_16x16x32_bf16(
            false, fa.v, false, fb.v, (short)0, acc, false, false);
    }

    // Epilogue: bias(es) + optional relu, scatter 8 rows of one column.
    const int crow0 = tileM + half * 8;
    const int ccol  = tileN + nlocal;
    float b = 0.0f;
    if (bias0) b += bias0[ccol];
    if (bias1) b += bias1[ccol];
#pragma unroll
    for (int r = 0; r < 8; ++r) {
        float v = acc[r] + b;
        if (do_relu) v = fmaxf(v, 0.0f);
        C[(size_t)(crow0 + r) * ldc + ccol] = v;
    }
}

// ---------------------------------------------------------------------------
// f32 -> bf16 cast with output leading dimension (lets us write the state
// slice straight into the concatenated [iw | state] GEMM operand).
// ---------------------------------------------------------------------------
__global__ void cast_f32_bf16_ld(const float* __restrict__ src, int M, int N,
                                 bf16_t* __restrict__ dst, int ldd)
{
    int idx = blockIdx.x * blockDim.x + threadIdx.x;
    if (idx >= M * N) return;
    int i = idx / N, j = idx - i * N;
    dst[(size_t)i * ldd + j] = (bf16_t)src[idx];
}

// ---------------------------------------------------------------------------
// LDS-tiled transpose + cast: WT[n*ldt + k] = bf16(W[k*N + n])
// Produces the K-major weight layout the GEMM's B-fragment loads want.
// ---------------------------------------------------------------------------
__global__ __launch_bounds__(256)
void transpose_cast_f32_bf16(const float* __restrict__ W, int K, int N,
                             bf16_t* __restrict__ WT, int ldt)
{
    __shared__ float tile[32][33];
    const int k0 = blockIdx.y * 32;
    const int n0 = blockIdx.x * 32;
    const int tx = threadIdx.x & 31;
    const int ty = threadIdx.x >> 5;           // 0..7
#pragma unroll
    for (int r = ty; r < 32; r += 8)
        tile[r][tx] = W[(size_t)(k0 + r) * N + (n0 + tx)];
    __syncthreads();
#pragma unroll
    for (int r = ty; r < 32; r += 8)
        WT[(size_t)(n0 + r) * ldt + (k0 + tx)] = (bf16_t)tile[tx][r];
}

// ---------------------------------------------------------------------------
// Fused Bahdanau score: att[i,j] = bf + sum_k relu(att1[j,k]+att2[i,k])*Wf[k]
// The (B,B,ATT) tensor is never materialized (saves >1 GB of HBM traffic).
// Block computes a 16x16 (i,j) tile; K streamed through LDS in chunks of 64.
// ---------------------------------------------------------------------------
__global__ __launch_bounds__(256)
void att_score(const float* __restrict__ att1,   // (BAG, ATT)  key rows j
               const float* __restrict__ att2,   // (BAG, ATT)  query rows i
               const float* __restrict__ Wf,     // (ATT)
               const float* __restrict__ bfp,    // (1)
               float* __restrict__ att)          // (BAG, BAG)
{
    __shared__ float sA1[16][64];
    __shared__ float sA2[16][64];
    __shared__ float sW[64];
    const int i0 = blockIdx.y * 16;
    const int j0 = blockIdx.x * 16;
    const int ti = threadIdx.x >> 4;   // query within tile
    const int tj = threadIdx.x & 15;   // key within tile
    float acc = 0.0f;
    for (int kc = 0; kc < ATT; kc += 64) {
        for (int e = threadIdx.x; e < 16 * 64; e += 256) {
            int rr = e >> 6, cc = e & 63;
            sA1[rr][cc] = att1[(size_t)(j0 + rr) * ATT + kc + cc];
            sA2[rr][cc] = att2[(size_t)(i0 + rr) * ATT + kc + cc];
        }
        if (threadIdx.x < 64) sW[threadIdx.x] = Wf[kc + threadIdx.x];
        __syncthreads();
#pragma unroll 8
        for (int k = 0; k < 64; ++k) {
            float h = sA1[tj][k] + sA2[ti][k];
            acc = fmaf(fmaxf(h, 0.0f), sW[k], acc);
        }
        __syncthreads();
    }
    att[(size_t)(i0 + ti) * BAG + (j0 + tj)] = acc + bfp[0];
}

// ---------------------------------------------------------------------------
// Row softmax (512 wide) -> bf16 alpha (feeds the awe WMMA GEMM directly).
// ---------------------------------------------------------------------------
__global__ __launch_bounds__(256)
void softmax_rows_bf16(const float* __restrict__ att, bf16_t* __restrict__ alphah)
{
    __shared__ float red[256];
    const float* a = att + (size_t)blockIdx.x * BAG;
    const int t = threadIdx.x;
    float v0 = a[t], v1 = a[t + 256];
    red[t] = fmaxf(v0, v1);
    __syncthreads();
    for (int s = 128; s > 0; s >>= 1) {
        if (t < s) red[t] = fmaxf(red[t], red[t + s]);
        __syncthreads();
    }
    float rowmax = red[0];
    __syncthreads();
    float e0 = __expf(v0 - rowmax), e1 = __expf(v1 - rowmax);
    red[t] = e0 + e1;
    __syncthreads();
    for (int s = 128; s > 0; s >>= 1) {
        if (t < s) red[t] += red[t + s];
        __syncthreads();
    }
    float inv = 1.0f / red[0];
    bf16_t* o = alphah + (size_t)blockIdx.x * BAG;
    o[t]       = (bf16_t)(e0 * inv);
    o[t + 256] = (bf16_t)(e1 * inv);
}

// ---------------------------------------------------------------------------
// input_weighted = x * awe, cast to bf16 into the concatenated operand.
// ---------------------------------------------------------------------------
__global__ void mul_cast_bf16(const float* __restrict__ x,
                              const float* __restrict__ awe,
                              bf16_t* __restrict__ dst, int ldd)
{
    int idx = blockIdx.x * blockDim.x + threadIdx.x;
    if (idx >= BAG * ENC) return;
    int i = idx >> 9, j = idx & 511;
    dst[(size_t)i * ldd + j] = (bf16_t)(x[idx] * awe[idx]);
}

// ---------------------------------------------------------------------------
// output = new_state @ W3 + b3  (N=2: wave-per-row dot + shfl reduction)
// ---------------------------------------------------------------------------
__global__ __launch_bounds__(256)
void final_fc(const float* __restrict__ ns, const float* __restrict__ W3,
              const float* __restrict__ b3, float* __restrict__ out)
{
    const int wave = threadIdx.x >> 5;
    const int lane = threadIdx.x & 31;
    const int row  = blockIdx.x * 8 + wave;
    const float* r = ns + (size_t)row * NDIMS;
    float s0 = 0.0f, s1 = 0.0f;
    for (int k = lane; k < NDIMS; k += 32) {
        float v = r[k];
        s0 = fmaf(v, W3[k * 2 + 0], s0);
        s1 = fmaf(v, W3[k * 2 + 1], s1);
    }
    for (int off = 16; off > 0; off >>= 1) {
        s0 += __shfl_down(s0, off, 32);
        s1 += __shfl_down(s1, off, 32);
    }
    if (lane == 0) {
        out[row * 2 + 0] = s0 + b3[0];
        out[row * 2 + 1] = s1 + b3[1];
    }
}

// ---------------------------------------------------------------------------
// Launch pipeline
// ---------------------------------------------------------------------------
extern "C" void kernel_launch(void* const* d_in, const int* in_sizes, int n_in,
                              void* d_out, int out_size, void* d_ws, size_t ws_size,
                              hipStream_t stream)
{
    const float* x     = (const float*)d_in[0];   // (512, 512)
    const float* state = (const float*)d_in[1];   // (512, 1024)
    const float* We    = (const float*)d_in[2];
    const float* be    = (const float*)d_in[3];
    const float* Wd    = (const float*)d_in[4];
    const float* bd    = (const float*)d_in[5];
    const float* Wf    = (const float*)d_in[6];
    const float* bfp   = (const float*)d_in[7];
    const float* W1    = (const float*)d_in[8];
    const float* b1    = (const float*)d_in[9];
    const float* W2    = (const float*)d_in[10];
    const float* b2    = (const float*)d_in[11];
    const float* W3    = (const float*)d_in[12];
    const float* b3    = (const float*)d_in[13];

    // Output: [output (512,2) | new_state (512,1024)] flat.
    float* out_logits = (float*)d_out;
    float* new_state  = (float*)d_out + BAG * 2;

    // Workspace carve-up (all sizes multiples of 16B).
    bf16_t* xh     = (bf16_t*)d_ws;                 // (512, 512)  A of att1
    bf16_t* xT     = xh     + BAG * ENC;            // (512, 512)  BT of awe
    bf16_t* WeT    = xT     + ENC * BAG;            // (512, 512)
    bf16_t* WdT    = WeT    + ATT * ENC;            // (512, 1024)
    bf16_t* alphah = WdT    + ATT * NDIMS;          // (512, 512)  A of awe
    bf16_t* Acat   = alphah + BAG * BAG;            // (512, 1536) [iw | state]
    bf16_t* WcatT  = Acat   + BAG * KCAT;           // (1024, 1536)
    float*  att1   = (float*)(WcatT + NDIMS * KCAT);
    float*  att2   = att1 + BAG * ATT;
    float*  attm   = att2 + BAG * ATT;
    float*  awe    = attm + BAG * BAG;
    (void)in_sizes; (void)n_in; (void)out_size; (void)ws_size;

    // --- operand preparation (cast / transpose to bf16, K-major weights) ---
    cast_f32_bf16_ld<<<(BAG * ENC) / 256, 256, 0, stream>>>(x, BAG, ENC, xh, ENC);
    cast_f32_bf16_ld<<<(BAG * NDIMS) / 256, 256, 0, stream>>>(state, BAG, NDIMS,
                                                              Acat + ENC, KCAT);
    transpose_cast_f32_bf16<<<dim3(BAG / 32, ENC / 32), 256, 0, stream>>>(x, BAG, ENC, xT, BAG);
    transpose_cast_f32_bf16<<<dim3(ATT / 32, ENC / 32), 256, 0, stream>>>(We, ENC, ATT, WeT, ENC);
    transpose_cast_f32_bf16<<<dim3(ATT / 32, NDIMS / 32), 256, 0, stream>>>(Wd, NDIMS, ATT, WdT, NDIMS);
    transpose_cast_f32_bf16<<<dim3(NDIMS / 32, ENC / 32), 256, 0, stream>>>(W1, ENC, NDIMS,
                                                                            WcatT, KCAT);
    transpose_cast_f32_bf16<<<dim3(NDIMS / 32, NDIMS / 32), 256, 0, stream>>>(W2, NDIMS, NDIMS,
                                                                              WcatT + ENC, KCAT);

    // --- att1 = x @ We + be ;  att2 = state @ Wd + bd  (WMMA bf16) ---
    gemm_bf16_wmma<<<dim3(ATT / 32, BAG / 64), 256, 0, stream>>>(
        xh, ENC, WeT, ENC, att1, ATT, ENC, be, nullptr, 0);
    gemm_bf16_wmma<<<dim3(ATT / 32, BAG / 64), 256, 0, stream>>>(
        Acat + ENC, KCAT, WdT, NDIMS, att2, ATT, NDIMS, bd, nullptr, 0);

    // --- fused relu(att1+att2) @ Wf  (h never materialized) ---
    att_score<<<dim3(BAG / 16, BAG / 16), 256, 0, stream>>>(att1, att2, Wf, bfp, attm);

    // --- softmax over keys -> bf16 alpha ---
    softmax_rows_bf16<<<BAG, 256, 0, stream>>>(attm, alphah);

    // --- awe = alpha @ x  (WMMA bf16) ---
    gemm_bf16_wmma<<<dim3(ENC / 32, BAG / 64), 256, 0, stream>>>(
        alphah, BAG, xT, BAG, awe, ENC, BAG, nullptr, nullptr, 0);

    // --- input_weighted = x * awe -> Acat[:, 0:512] ---
    mul_cast_bf16<<<(BAG * ENC) / 256, 256, 0, stream>>>(x, awe, Acat, KCAT);

    // --- new_state = relu([iw|state] @ [W1;W2] + b1 + b2)  (single WMMA GEMM) ---
    gemm_bf16_wmma<<<dim3(NDIMS / 32, BAG / 64), 256, 0, stream>>>(
        Acat, KCAT, WcatT, KCAT, new_state, NDIMS, KCAT, b1, b2, 1);

    // --- output = new_state @ W3 + b3 ---
    final_fc<<<BAG / 8, 256, 0, stream>>>(new_state, W3, b3, out_logits);
}
